// Attention_10471130268169
// MI455X (gfx1250) — matrix operand order repaired and checked by reading, not compiled, not run on hardware
//
#include <hip/hip_runtime.h>

// ---------------------------------------------------------------------------
// MI455X (gfx1250) fused multi-head attention, bf16 WMMA everywhere.
// ~137 GFLOP vs ~150 MB HBM traffic -> matrix-compute bound: every GEMM uses
// v_wmma_f32_16x16x32_bf16, logits never hit HBM (flash softmax), all LDS
// operand tiles give each lane contiguous 16B-aligned fragments
// (ds_load_b128), global->LDS staging is double-buffered with next-tile
// loads issued AFTER the barrier (so they overlap the WMMA burst), and
// row-major tile copies go through the Tensor Data Mover.
// ---------------------------------------------------------------------------

typedef __attribute__((ext_vector_type(16))) __bf16 v16bf;
typedef __attribute__((ext_vector_type(8)))  __bf16 v8bf;
typedef __attribute__((ext_vector_type(4)))  __bf16 v4bf;
typedef __attribute__((ext_vector_type(8)))  float  v8f;

#define WMMA_BF16(a, b, c) \
  __builtin_amdgcn_wmma_f32_16x16x32_bf16(false, (a), false, (b), (short)0, (c), false, false)

#if defined(__has_builtin)
#if __has_builtin(__builtin_amdgcn_tensor_load_to_lds) && \
    __has_builtin(__builtin_amdgcn_s_wait_tensorcnt)
#define USE_TDM 1
#endif
#endif
#ifndef USE_TDM
#define USE_TDM 0
#endif

// Combine two 16B chunks into a 32B fragment; guarantees 2 vector loads.
__device__ __forceinline__ v16bf ld_frag(const __bf16* p0, const __bf16* p1) {
  v8bf lo = *reinterpret_cast<const v8bf*>(p0);
  v8bf hi = *reinterpret_cast<const v8bf*>(p1);
  return __builtin_shufflevector(lo, hi, 0, 1, 2, 3, 4, 5, 6, 7,
                                 8, 9, 10, 11, 12, 13, 14, 15);
}

#if USE_TDM
typedef unsigned int v4u_ __attribute__((ext_vector_type(4)));
typedef int          v8i_ __attribute__((ext_vector_type(8)));
typedef int          v4i_ __attribute__((ext_vector_type(4)));

__device__ __forceinline__ unsigned int lds_off(const void* p) {
  // flat shared address truncates to the LDS byte offset (ISA 10.2 aperture)
  return (unsigned int)(unsigned long long)p;
}

// 2D bf16 tile load via Tensor Data Mover. rows x row_elems tile, row stride
// in elements; LDS destination gets pad_amount (code) dwords of padding every
// pad_interval (code) dwords to reproduce padded LDS row strides.
__device__ __forceinline__ void tdm_load_2d_bf16(const __bf16* gsrc,
                                                 unsigned int lds_byte,
                                                 unsigned int rows,
                                                 unsigned int row_elems,
                                                 unsigned int row_stride,
                                                 unsigned int pad_int_code,
                                                 unsigned int pad_amt_code) {
  unsigned long long ga = (unsigned long long)gsrc;
  v4u_ g0;
  g0[0] = 1u;                                           // count=1 (valid user D#)
  g0[1] = lds_byte;                                     // lds_addr
  g0[2] = (unsigned int)ga;                             // global_addr[31:0]
  g0[3] = (unsigned int)((ga >> 32) & 0x01ffffffu) | (2u << 30);  // addr hi | type=2
  v8i_ g1;
  g1[0] = (int)((1u << 16) |                            // data_size = 2B
                (1u << 20) |                            // pad_enable
                (pad_int_code << 22) | (pad_amt_code << 25));
  g1[1] = (int)(row_elems << 16);                       // tensor_dim0[15:0]
  g1[2] = (int)(rows << 16);                            // tensor_dim1[15:0]
  g1[3] = (int)(row_elems << 16);                       // tile_dim0
  g1[4] = (int)rows;                                    // tile_dim1 (tile_dim2=0)
  g1[5] = (int)row_stride;                              // tensor_dim0_stride lo
  g1[6] = 0;
  g1[7] = 0;
  v4i_ z4 = {0, 0, 0, 0};
#if __clang_major__ >= 23
  v8i_ z8 = {0, 0, 0, 0, 0, 0, 0, 0};
  __builtin_amdgcn_tensor_load_to_lds(g0, g1, z4, z4, z8, 0);
#else
  __builtin_amdgcn_tensor_load_to_lds(g0, g1, z4, z4, 0);
#endif
}
#endif  // USE_TDM

// Problem constants: B=4, S=2048, D=1024, H=16, Dh=64
constexpr int SEQ = 2048;
constexpr int NB  = 4;
constexpr int NH  = 16;
constexpr int DH  = 64;
constexpr int DM  = 1024;          // model dim = H*Dh = D
constexpr int MROWS = NB * SEQ;    // 8192

// ---------------------------------------------------------------------------
__global__ __launch_bounds__(256) void cvt_f32_bf16(const float* __restrict__ in,
                                                    __bf16* __restrict__ out, int n) {
  int i = blockIdx.x * 256 + threadIdx.x;
  if (i < n) out[i] = (__bf16)in[i];
}

// ---------------------------------------------------------------------------
// GEMM: A[f32 MxK] * Bw[bf16 KxN] -> C[bf16 MxN] * scale.  Tile 256x64,
// 8 waves x (32x64). A converted f32->bf16 while staging (manual path);
// B staged transposed Bt[n][k]. Double-buffered k-loop; next-tile global
// loads issued after the barrier so they overlap this tile's WMMAs.
// ---------------------------------------------------------------------------
__global__ __launch_bounds__(256) void gemm_f32a_bf16o(const float* __restrict__ A,
                                                       const __bf16* __restrict__ Bw,
                                                       __bf16* __restrict__ C,
                                                       int M, int N, int K, float scale) {
  constexpr int ALDA = 40;   // 32 + pad halves (80B rows)
  constexpr int BTP  = 40;
  __shared__ __align__(16) __bf16 Alds[2][256 * ALDA];
  __shared__ __align__(16) __bf16 Bt[2][64 * BTP];

  const int tid = threadIdx.x, lane = tid & 31, wv = tid >> 5;
  const int ln = lane & 15, lh = lane >> 4;
  const int m0 = blockIdx.x * 256, n0 = blockIdx.y * 64;

  v8f acc[8] = {};
  float4 fa[8];
  v8bf fb;

  auto gload = [&](int kb) {
#pragma unroll
    for (int i = 0; i < 8; ++i) {
      int c = tid + i * 256;
      fa[i] = *reinterpret_cast<const float4*>(
          &A[(size_t)(m0 + (c >> 3)) * K + kb + (c & 7) * 4]);
    }
    fb = *reinterpret_cast<const v8bf*>(
        &Bw[(size_t)(kb + (tid >> 3)) * N + n0 + (tid & 7) * 8]);
  };
  auto lstore = [&](int buf) {
#pragma unroll
    for (int i = 0; i < 8; ++i) {
      int c = tid + i * 256, row = c >> 3, col = (c & 7) * 4;
      v4bf t;
      t[0] = (__bf16)fa[i].x; t[1] = (__bf16)fa[i].y;
      t[2] = (__bf16)fa[i].z; t[3] = (__bf16)fa[i].w;
      *reinterpret_cast<v4bf*>(&Alds[buf][row * ALDA + col]) = t;
    }
    int row = tid >> 3, col = (tid & 7) * 8;
#pragma unroll
    for (int i = 0; i < 8; ++i) Bt[buf][(col + i) * BTP + row] = fb[i];
  };

  gload(0);
  lstore(0);
  int cur = 0;
  for (int kb = 0; kb < K; kb += 32) {
    const bool nxt = (kb + 32) < K;
    __syncthreads();                     // publish buffer `cur`
    if (nxt) gload(kb + 32);             // issued AFTER barrier: overlaps WMMAs
    const __bf16* pa0 = &Alds[cur][(wv * 32 + ln) * ALDA + lh * 8];
    const __bf16* pa1 = pa0 + 16 * ALDA;
    v16bf a0 = ld_frag(pa0, pa0 + 16);
    v16bf a1 = ld_frag(pa1, pa1 + 16);
#pragma unroll
    for (int j = 0; j < 4; ++j) {
      const __bf16* pb = &Bt[cur][(j * 16 + ln) * BTP + lh * 8];
      v16bf b = ld_frag(pb, pb + 16);
      acc[j]     = WMMA_BF16(a0, b, acc[j]);
      acc[4 + j] = WMMA_BF16(a1, b, acc[4 + j]);
    }
    if (nxt) lstore(cur ^ 1);
    cur ^= 1;
  }

#pragma unroll
  for (int i = 0; i < 2; ++i)
#pragma unroll
    for (int j = 0; j < 4; ++j)
#pragma unroll
      for (int r = 0; r < 8; ++r) {
        int rowg = m0 + wv * 32 + i * 16 + r + lh * 8;
        int colg = n0 + j * 16 + ln;
        C[(size_t)rowg * N + colg] = (__bf16)(acc[i * 4 + j][r] * scale);
      }
}

// ---------------------------------------------------------------------------
// GEMM: A[bf16 MxK] * Bw[bf16 KxN] -> C[f32 MxN].  A tile is a plain
// row-major copy -> staged by the Tensor Data Mover when available.
// ---------------------------------------------------------------------------
__global__ __launch_bounds__(256) void gemm_bf16a_f32o(const __bf16* __restrict__ A,
                                                       const __bf16* __restrict__ Bw,
                                                       float* __restrict__ C,
                                                       int M, int N, int K) {
  constexpr int ALDA = 40;
  constexpr int BTP  = 40;
  __shared__ __align__(16) __bf16 Alds[2][256 * ALDA];
  __shared__ __align__(16) __bf16 Bt[2][64 * BTP];

  const int tid = threadIdx.x, lane = tid & 31, wv = tid >> 5;
  const int ln = lane & 15, lh = lane >> 4;
  const int m0 = blockIdx.x * 256, n0 = blockIdx.y * 64;

  v8f acc[8] = {};
  v8bf fb;
#if !USE_TDM
  v8bf fA[4];
#endif

  auto gloadB = [&](int kb) {
    fb = *reinterpret_cast<const v8bf*>(
        &Bw[(size_t)(kb + (tid >> 3)) * N + n0 + (tid & 7) * 8]);
  };
  auto lstoreB = [&](int buf) {
    int row = tid >> 3, col = (tid & 7) * 8;
#pragma unroll
    for (int i = 0; i < 8; ++i) Bt[buf][(col + i) * BTP + row] = fb[i];
  };
#if USE_TDM
  auto tdmA = [&](int kb, int buf) {
    if (wv == 0)
      tdm_load_2d_bf16(&A[(size_t)m0 * K + kb], lds_off(&Alds[buf][0]),
                       /*rows=*/256, /*row_elems=*/32, /*stride=*/(unsigned)K,
                       /*pad_interval: 16 dwords=*/3, /*pad: 4 dwords=*/2);
  };
#else
  auto gloadA = [&](int kb) {
#pragma unroll
    for (int i = 0; i < 4; ++i) {
      int c = tid + i * 256;
      fA[i] = *reinterpret_cast<const v8bf*>(
          &A[(size_t)(m0 + (c >> 2)) * K + kb + (c & 3) * 8]);
    }
  };
  auto lstoreA = [&](int buf) {
#pragma unroll
    for (int i = 0; i < 4; ++i) {
      int c = tid + i * 256, row = c >> 2, col = (c & 3) * 8;
      *reinterpret_cast<v8bf*>(&Alds[buf][row * ALDA + col]) = fA[i];
    }
  };
#endif

#if USE_TDM
  tdmA(0, 0);
#else
  gloadA(0);
  lstoreA(0);
#endif
  gloadB(0);
  lstoreB(0);

  int cur = 0;
  for (int kb = 0; kb < K; kb += 32) {
    const bool nxt = (kb + 32) < K;
#if USE_TDM
    if (wv == 0) __builtin_amdgcn_s_wait_tensorcnt((short)0);
#endif
    __syncthreads();                     // buffer `cur` (TDM + DS stores) visible
    if (nxt) {
#if USE_TDM
      tdmA(kb + 32, cur ^ 1);            // async DMA into back buffer
#else
      gloadA(kb + 32);
#endif
      gloadB(kb + 32);
    }
    const __bf16* pa0 = &Alds[cur][(wv * 32 + ln) * ALDA + lh * 8];
    const __bf16* pa1 = pa0 + 16 * ALDA;
    v16bf a0 = ld_frag(pa0, pa0 + 16);
    v16bf a1 = ld_frag(pa1, pa1 + 16);
#pragma unroll
    for (int j = 0; j < 4; ++j) {
      const __bf16* pb = &Bt[cur][(j * 16 + ln) * BTP + lh * 8];
      v16bf b = ld_frag(pb, pb + 16);
      acc[j]     = WMMA_BF16(a0, b, acc[j]);
      acc[4 + j] = WMMA_BF16(a1, b, acc[4 + j]);
    }
    if (nxt) {
#if !USE_TDM
      lstoreA(cur ^ 1);
#endif
      lstoreB(cur ^ 1);
    }
    cur ^= 1;
  }

#pragma unroll
  for (int i = 0; i < 2; ++i)
#pragma unroll
    for (int j = 0; j < 4; ++j)
#pragma unroll
      for (int r = 0; r < 8; ++r) {
        int rowg = m0 + wv * 32 + i * 16 + r + lh * 8;
        int colg = n0 + j * 16 + ln;
        C[(size_t)rowg * N + colg] = acc[i * 4 + j][r];
      }
}

// ---------------------------------------------------------------------------
// Flash attention forward (Q pre-scaled by 1/sqrt(Dh)).
// Grid (S/128, B*H); 8 waves, 16 queries/wave; K/V streamed in 32-key blocks,
// double-buffered. K tile row-major (TDM copy when available); V transposed
// manually (TDM cannot transpose 2B elements).
// ---------------------------------------------------------------------------
__global__ __launch_bounds__(256) void attn_fwd(const __bf16* __restrict__ Q,
                                                const __bf16* __restrict__ Km,
                                                const __bf16* __restrict__ Vm,
                                                __bf16* __restrict__ Ctx) {
  constexpr int LDK = 80;   // 64 + 16 pad halves (TDM: interval 32dw, pad 8dw)
  constexpr int VTP = 40;
  constexpr int LDP = 40;
  __shared__ __align__(16) __bf16 Klds[2][32 * LDK];
  __shared__ __align__(16) __bf16 Vt[2][64 * VTP];
  __shared__ __align__(16) __bf16 Plds[8 * 16 * LDP];

  const int tid = threadIdx.x, lane = tid & 31, wv = tid >> 5;
  const int ln = lane & 15, lh = lane >> 4;
  const int bh = blockIdx.y;            // b*H + h
  const int b = bh >> 4, h = bh & 15;
  const size_t rs   = DM;
  const size_t base = (size_t)b * SEQ * rs + (size_t)h * DH;
  const int q0 = blockIdx.x * 128 + wv * 16;

  const __bf16* qp = &Q[base + (size_t)(q0 + ln) * rs + lh * 8];
  v16bf aq0 = ld_frag(qp, qp + 16);
  v16bf aq1 = ld_frag(qp + 32, qp + 48);

  float mrow[8], lrow[8];
#pragma unroll
  for (int r = 0; r < 8; ++r) { mrow[r] = -3.0e38f; lrow[r] = 0.0f; }
  v8f acc[4] = {};
  __bf16* Pw = &Plds[wv * 16 * LDP];

  const int srow = tid >> 3, scol = (tid & 7) * 8;   // staging coords

  // prologue: stage block 0 into buffer 0
  {
    size_t g = base + (size_t)srow * rs + scol;
#if USE_TDM
    if (wv == 0)
      tdm_load_2d_bf16(&Km[base], lds_off(&Klds[0][0]), 32, 64, DM,
                       /*interval 32dw=*/4, /*pad 8dw=*/7);
#else
    *reinterpret_cast<v8bf*>(&Klds[0][srow * LDK + scol]) =
        *reinterpret_cast<const v8bf*>(&Km[g]);
#endif
    v8bf s = *reinterpret_cast<const v8bf*>(&Vm[g]);
#pragma unroll
    for (int i = 0; i < 8; ++i) Vt[0][(scol + i) * VTP + srow] = s[i];
  }

  int cur = 0;
  for (int tb = 0; tb < SEQ; tb += 32) {
    const bool nxt = (tb + 32) < SEQ;
#if USE_TDM
    if (wv == 0) __builtin_amdgcn_s_wait_tensorcnt((short)0);
#endif
    __syncthreads();                    // buffer `cur` visible to all waves

    v8bf vreg;
#if !USE_TDM
    v8bf kreg;
#endif
    if (nxt) {
      size_t g = base + (size_t)(tb + 32 + srow) * rs + scol;
#if USE_TDM
      if (wv == 0)
        tdm_load_2d_bf16(&Km[base + (size_t)(tb + 32) * rs],
                         lds_off(&Klds[cur ^ 1][0]), 32, 64, DM, 4, 7);
#else
      kreg = *reinterpret_cast<const v8bf*>(&Km[g]);
#endif
      vreg = *reinterpret_cast<const v8bf*>(&Vm[g]);
    }

    // logits S = Q*K^T : 16x32 as two 16x16 accum tiles
    v8f c0 = {}, c1 = {};
    {
      const __bf16* kp0 = &Klds[cur][ln * LDK + lh * 8];
      const __bf16* kp1 = &Klds[cur][(16 + ln) * LDK + lh * 8];
      c0 = WMMA_BF16(aq0, ld_frag(kp0, kp0 + 16), c0);
      c0 = WMMA_BF16(aq1, ld_frag(kp0 + 32, kp0 + 48), c0);
      c1 = WMMA_BF16(aq0, ld_frag(kp1, kp1 + 16), c1);
      c1 = WMMA_BF16(aq1, ld_frag(kp1 + 32, kp1 + 48), c1);
    }

    // online softmax; row f = r + lh*8 spans the 16 lanes of a half
#pragma unroll
    for (int r = 0; r < 8; ++r) {
      float mx = fmaxf(c0[r], c1[r]);
      mx = fmaxf(mx, __shfl_xor(mx, 1, 32));
      mx = fmaxf(mx, __shfl_xor(mx, 2, 32));
      mx = fmaxf(mx, __shfl_xor(mx, 4, 32));
      mx = fmaxf(mx, __shfl_xor(mx, 8, 32));
      float mnew  = fmaxf(mrow[r], mx);
      float p0    = __expf(c0[r] - mnew);
      float p1    = __expf(c1[r] - mnew);
      float alpha = __expf(mrow[r] - mnew);
      float sum = p0 + p1;
      sum += __shfl_xor(sum, 1, 32);
      sum += __shfl_xor(sum, 2, 32);
      sum += __shfl_xor(sum, 4, 32);
      sum += __shfl_xor(sum, 8, 32);
      lrow[r] = lrow[r] * alpha + sum;
      mrow[r] = mnew;
      acc[0][r] *= alpha; acc[1][r] *= alpha; acc[2][r] *= alpha; acc[3][r] *= alpha;
      Pw[(r + lh * 8) * LDP + ln]      = (__bf16)p0;
      Pw[(r + lh * 8) * LDP + 16 + ln] = (__bf16)p1;
    }
    // intra-wave cross-lane LDS visibility: DS-counter wait, no full barrier
    asm volatile("s_wait_dscnt 0" ::: "memory");

    const __bf16* pp = &Pw[ln * LDP + lh * 8];
    v16bf pa = ld_frag(pp, pp + 16);    // P as 16x32 A fragment
#pragma unroll
    for (int j = 0; j < 4; ++j) {
      const __bf16* vp = &Vt[cur][(j * 16 + ln) * VTP + lh * 8];
      acc[j] = WMMA_BF16(pa, ld_frag(vp, vp + 16), acc[j]);
    }

    if (nxt) {
#if !USE_TDM
      *reinterpret_cast<v8bf*>(&Klds[cur ^ 1][srow * LDK + scol]) = kreg;
#endif
#pragma unroll
      for (int i = 0; i < 8; ++i) Vt[cur ^ 1][(scol + i) * VTP + srow] = vreg[i];
    }
    cur ^= 1;
  }

  // normalize and store ctx (bf16)
#pragma unroll
  for (int r = 0; r < 8; ++r) {
    float inv = 1.0f / lrow[r];
    size_t rowg = base + (size_t)(q0 + r + lh * 8) * rs;
#pragma unroll
    for (int j = 0; j < 4; ++j)
      Ctx[rowg + j * 16 + ln] = (__bf16)(acc[j][r] * inv);
  }
}

// ---------------------------------------------------------------------------
// launch
// ---------------------------------------------------------------------------
extern "C" void kernel_launch(void* const* d_in, const int* in_sizes, int n_in,
                              void* d_out, int out_size, void* d_ws, size_t ws_size,
                              hipStream_t stream) {
  (void)in_sizes; (void)n_in; (void)out_size; (void)ws_size;
  const float* Xq = (const float*)d_in[0];
  const float* Xk = (const float*)d_in[1];
  const float* Xv = (const float*)d_in[2];
  const float* Wq = (const float*)d_in[3];
  const float* Wk = (const float*)d_in[4];
  const float* Wv = (const float*)d_in[5];
  const float* Wo = (const float*)d_in[6];
  float* out = (float*)d_out;

  char* ws = (char*)d_ws;
  size_t off = 0;
  auto carve = [&](size_t bytes) {
    void* p = ws + off;
    off += (bytes + 255) & ~(size_t)255;
    return p;
  };
  const size_t wbytes = (size_t)DM * DM * sizeof(__bf16);      // 2 MB each
  const size_t abytes = (size_t)MROWS * DM * sizeof(__bf16);   // 16 MB each
  __bf16* Wqb = (__bf16*)carve(wbytes);
  __bf16* Wkb = (__bf16*)carve(wbytes);
  __bf16* Wvb = (__bf16*)carve(wbytes);
  __bf16* Wob = (__bf16*)carve(wbytes);
  __bf16* Qb  = (__bf16*)carve(abytes);
  __bf16* Kb  = (__bf16*)carve(abytes);
  __bf16* Vb  = (__bf16*)carve(abytes);
  __bf16* Cb  = (__bf16*)carve(abytes);

  const int wn = DM * DM;
  cvt_f32_bf16<<<wn / 256, 256, 0, stream>>>(Wq, Wqb, wn);
  cvt_f32_bf16<<<wn / 256, 256, 0, stream>>>(Wk, Wkb, wn);
  cvt_f32_bf16<<<wn / 256, 256, 0, stream>>>(Wv, Wvb, wn);
  cvt_f32_bf16<<<wn / 256, 256, 0, stream>>>(Wo, Wob, wn);

  dim3 gg(MROWS / 256, DM / 64);   // 32 x 16
  gemm_f32a_bf16o<<<gg, 256, 0, stream>>>(Xq, Wqb, Qb, MROWS, DM, DM, 0.125f); // fold 1/sqrt(Dh)
  gemm_f32a_bf16o<<<gg, 256, 0, stream>>>(Xk, Wkb, Kb, MROWS, DM, DM, 1.0f);
  gemm_f32a_bf16o<<<gg, 256, 0, stream>>>(Xv, Wvb, Vb, MROWS, DM, DM, 1.0f);

  attn_fwd<<<dim3(SEQ / 128, NB * NH), 256, 0, stream>>>(Qb, Kb, Vb, Cb);

  gemm_bf16a_f32o<<<gg, 256, 0, stream>>>(Cb, Wob, out, MROWS, DM, DM);
}
